// DecoderTransformerBackbone_1589137900084
// MI455X (gfx1250) — matrix operand order repaired and checked by reading, not compiled, not run on hardware
//
#include <hip/hip_runtime.h>

typedef __bf16 bf16_t;
typedef bf16_t v16bf __attribute__((ext_vector_type(16)));
typedef bf16_t v8bf  __attribute__((ext_vector_type(8)));
typedef float  v8f   __attribute__((ext_vector_type(8)));

enum {
    GF_RELU      = 1,
    GF_MASK      = 2,   // causal averaging mask: *(1/(i+1)) if j<=i else 0
    GF_BIAS      = 4,
    GF_RESIDUAL  = 8,
    GF_CAUSALK   = 16,  // K-reduction truncated at m0+BM (attn*V with causal S)
    GF_STORE_BFT = 32,  // store Cbf transposed: Cbf[j*M + i]
};

#define BM 64
#define BNT 128
#define BK 32
#define LDA_S 40   // 32 + 8 pad (80B row stride, 16B multiple)
#define LDB2  40   // Bs[n][k]: 32 k + 8 pad

// gfx1250 async global->LDS copy (ASYNCcnt); LDS generic addr low 32 bits are
// the DS byte offset (ISA 10.2 aperture rules).
__device__ __forceinline__ void async_b128(bf16_t* lds_dst, const bf16_t* gsrc)
{
    unsigned l = (unsigned)(uintptr_t)lds_dst;
    asm volatile("global_load_async_to_lds_b128 %0, %1, off"
                 :: "v"(l), "v"(gsrc)
                 : "memory");
}
__device__ __forceinline__ void wait_async()
{
    asm volatile("s_wait_asynccnt 0x0" ::: "memory");
}

// ---------------------------------------------------------------------------
// bf16 WMMA GEMM: C[M,N] = epilogue(A[M,K] * Bt[N,K]^T)
// A row-major [M][K]; B ALWAYS given transposed as Bt[N][K] so that staging
// into the fragment-ready LDS layout Bs[n][k] is a pure async vector copy.
// 128 threads = 4 waves; block tile 64x128; each wave 32x64 (2x4 WMMA).
// Double-buffered LDS, fully async global->LDS staging.
// Requires M%64==0, N%128==0, K%32==0 (true for all call sites here).
// ---------------------------------------------------------------------------
__global__ __launch_bounds__(128)
void gemm_bf16_wmma(const bf16_t* __restrict__ A, const bf16_t* __restrict__ Bt,
                    const float* __restrict__ bias, const float* __restrict__ resid,
                    float* __restrict__ Cf, bf16_t* __restrict__ Cbf,
                    int M, int N, int Kd, int flags)
{
    __shared__ bf16_t As[2][BM * LDA_S];    // 10 KB
    __shared__ bf16_t Bs[2][BNT * LDB2];    // 20 KB  (fragment-ready: Bs[n][k])

    const int tid  = threadIdx.x;
    const int lane = tid & 31;
    const int wave = tid >> 5;
    const int half = lane >> 4;   // wave32 lane-half
    const int l16  = lane & 15;

    const int m0 = blockIdx.y * BM;
    const int n0 = blockIdx.x * BNT;
    const int wm = (wave >> 1) * 32;
    const int wn = (wave & 1) * 64;

    v8f acc[2][4] = {};

    const bool fullyMasked = (flags & GF_MASK) && (n0 > m0 + (BM - 1));

    int kend = Kd;
    if (flags & GF_CAUSALK) {
        int lim = m0 + BM;                       // need k < m0+BM
        if (lim < Kd) kend = ((lim + BK - 1) / BK) * BK;
    }

    auto stage = [&](int k0, int buf) {
        // A tile 64x32 : 256 v8bf slots over 128 threads (async)
        #pragma unroll
        for (int p = 0; p < 2; ++p) {
            int s = tid + p * 128;
            int row = s >> 2, c = s & 3;
            async_b128(&As[buf][row * LDA_S + c * 8],
                       &A[(size_t)(m0 + row) * Kd + k0 + c * 8]);
        }
        // B tile 128x32 from Bt[N][K]: direct async vector copy
        #pragma unroll
        for (int p = 0; p < 4; ++p) {
            int s = tid + p * 128;
            int n = s >> 2, c = s & 3;
            async_b128(&Bs[buf][n * LDB2 + c * 8],
                       &Bt[(size_t)(n0 + n) * Kd + k0 + c * 8]);
        }
    };

    auto compute = [&](int buf) {
        // B fragments: lane = column, K contiguous -> ds_load_b128 pairs
        v16bf bfr[4];
        #pragma unroll
        for (int bb = 0; bb < 4; ++bb) {
            int col = wn + bb * 16 + l16;
            v8bf lo = *(const v8bf*)&Bs[buf][col * LDB2 + half * 16];
            v8bf hi = *(const v8bf*)&Bs[buf][col * LDB2 + half * 16 + 8];
            bfr[bb] = __builtin_shufflevector(lo, hi,
                0, 1, 2, 3, 4, 5, 6, 7, 8, 9, 10, 11, 12, 13, 14, 15);
        }
        // A fragments (16x32 bf16: per-lane-half contiguous 16-k block) + WMMA
        #pragma unroll
        for (int a = 0; a < 2; ++a) {
            int row = wm + a * 16 + l16;
            v8bf lo = *(const v8bf*)&As[buf][row * LDA_S + half * 16];
            v8bf hi = *(const v8bf*)&As[buf][row * LDA_S + half * 16 + 8];
            v16bf af = __builtin_shufflevector(lo, hi,
                0, 1, 2, 3, 4, 5, 6, 7, 8, 9, 10, 11, 12, 13, 14, 15);
            #pragma unroll
            for (int bb = 0; bb < 4; ++bb) {
                acc[a][bb] = __builtin_amdgcn_wmma_f32_16x16x32_bf16(
                    false, af, false, bfr[bb], (short)0, acc[a][bb], false, false);
            }
        }
    };

    if (!fullyMasked) {
        stage(0, 0);
        wait_async();
        __syncthreads();
        int cur = 0;
        for (int k0 = 0; k0 < kend; k0 += BK) {
            int kn = k0 + BK;
            if (kn < kend) stage(kn, cur ^ 1);
            compute(cur);
            wait_async();
            __syncthreads();
            cur ^= 1;
        }
    }

    // ---- epilogue: C layout VGPR v -> M = v + 8*half ; N = lane&15
    #pragma unroll
    for (int a = 0; a < 2; ++a) {
        #pragma unroll
        for (int bb = 0; bb < 4; ++bb) {
            #pragma unroll
            for (int v = 0; v < 8; ++v) {
                int gi = m0 + wm + a * 16 + half * 8 + v;
                int gj = n0 + wn + bb * 16 + l16;
                float val = acc[a][bb][v];
                if (flags & GF_BIAS)     val += bias[gj];
                if (flags & GF_RELU)     val = val > 0.f ? val : 0.f;
                if (flags & GF_MASK)     val = (gj <= gi) ? val / (float)(gi + 1) : 0.f;
                if (flags & GF_RESIDUAL) val += resid[(size_t)gi * N + gj];
                if (Cf)  Cf[(size_t)gi * N + gj] = val;
                if (Cbf) {
                    size_t offb = (flags & GF_STORE_BFT) ? ((size_t)gj * M + gi)
                                                         : ((size_t)gi * N + gj);
                    Cbf[offb] = (bf16_t)val;
                }
            }
        }
    }
}

// ---------------------------------------------------------------------------
__global__ void add_pe_kernel(const float* __restrict__ x, const float* __restrict__ wpe,
                              float* __restrict__ H, bf16_t* __restrict__ Hbf,
                              int total, int ndmod)
{
    int idx = blockIdx.x * blockDim.x + threadIdx.x;
    if (idx >= total) return;
    float v = x[idx] + wpe[idx % ndmod];
    H[idx]   = v;
    Hbf[idx] = (bf16_t)v;
}

// fp32 [L][K][N] -> bf16 transposed [L][N][K] (weights, once per call)
__global__ void wT_to_bf16_kernel(const float* __restrict__ src, bf16_t* __restrict__ dst,
                                  int total, int D)
{
    int idx = blockIdx.x * blockDim.x + threadIdx.x;
    if (idx >= total) return;
    int dd = D * D;
    int l = idx / dd, r = idx % dd;
    int n = r / D, k = r % D;
    dst[idx] = (bf16_t)src[(size_t)l * dd + (size_t)k * D + n];
}

// ---------------------------------------------------------------------------
// Row LayerNorm over D=256 (one block of 256 threads per row); writes f32+bf16
// ---------------------------------------------------------------------------
__global__ __launch_bounds__(256)
void ln_kernel(const float* __restrict__ in, const float* __restrict__ g,
               const float* __restrict__ b, float* __restrict__ out,
               bf16_t* __restrict__ outbf, int D)
{
    __shared__ float red[256];
    const int row = blockIdx.x;
    const int t   = threadIdx.x;
    float v = in[(size_t)row * D + t];

    red[t] = v;
    __syncthreads();
    #pragma unroll
    for (int s = 128; s > 0; s >>= 1) {
        if (t < s) red[t] += red[t + s];
        __syncthreads();
    }
    float mu = red[0] / (float)D;
    __syncthreads();

    float dv = v - mu;
    red[t] = dv * dv;
    __syncthreads();
    #pragma unroll
    for (int s = 128; s > 0; s >>= 1) {
        if (t < s) red[t] += red[t + s];
        __syncthreads();
    }
    float var = red[0] / (float)D;
    float r = rsqrtf(var + 1e-5f);
    float o = dv * r * g[t] + b[t];
    out[(size_t)row * D + t]   = o;
    outbf[(size_t)row * D + t] = (bf16_t)o;
}

__global__ void copy_f32_kernel(const float* __restrict__ src, float* __restrict__ dst, int n)
{
    int idx = blockIdx.x * blockDim.x + threadIdx.x;
    if (idx < n) dst[idx] = src[idx];
}

// ---------------------------------------------------------------------------
extern "C" void kernel_launch(void* const* d_in, const int* in_sizes, int n_in,
                              void* d_out, int out_size, void* d_ws, size_t ws_size,
                              hipStream_t stream)
{
    const int Bb = 8, Nn = 2048, Dd = 256, Ll = 4;
    const int BN = Bb * Nn;              // 16384 rows
    const size_t HD  = (size_t)BN * Dd;  // 4,194,304 elements
    const size_t WSZ = (size_t)Ll * Dd * Dd;

    const float* x      = (const float*)d_in[0];
    const float* wpe    = (const float*)d_in[1];
    const float* Wq     = (const float*)d_in[2];
    const float* Wk     = (const float*)d_in[3];
    const float* Wv     = (const float*)d_in[4];
    const float* ln1_g  = (const float*)d_in[5];
    const float* ln1_b  = (const float*)d_in[6];
    const float* W1     = (const float*)d_in[7];
    const float* b1     = (const float*)d_in[8];
    const float* W2     = (const float*)d_in[9];
    const float* b2     = (const float*)d_in[10];
    const float* ln2_g  = (const float*)d_in[11];
    const float* ln2_b  = (const float*)d_in[12];

    char* p = (char*)d_ws;
    auto carve = [&](size_t bytes) -> void* {
        void* r = (void*)p;
        p += (bytes + 255) & ~(size_t)255;
        return r;
    };
    float*  H    = (float*) carve(HD * 4);
    bf16_t* Hbf  = (bf16_t*)carve(HD * 2);
    bf16_t* Qbf  = (bf16_t*)carve(HD * 2);
    bf16_t* Kbf  = (bf16_t*)carve(HD * 2);
    bf16_t* VtBf = (bf16_t*)carve(HD * 2);                // per batch: [D][N] (transposed)
    bf16_t* Sbf  = (bf16_t*)carve((size_t)Nn * Nn * 2);   // one batch of scores, reused
    float*  Tmp  = (float*) carve(HD * 4);
    bf16_t* h1bf = (bf16_t*)carve(HD * 2);
    bf16_t* WqT  = (bf16_t*)carve(WSZ * 2);               // all weights stored [N][K]
    bf16_t* WkT  = (bf16_t*)carve(WSZ * 2);
    bf16_t* WvT  = (bf16_t*)carve(WSZ * 2);
    bf16_t* W1T  = (bf16_t*)carve(WSZ * 2);
    bf16_t* W2T  = (bf16_t*)carve(WSZ * 2);

    // ---- convert + transpose weights to bf16 [L][N][K] (once per call)
    {
        int n = (int)WSZ, blk = 256, grd = (n + blk - 1) / blk;
        wT_to_bf16_kernel<<<grd, blk, 0, stream>>>(Wq, WqT, n, Dd);
        wT_to_bf16_kernel<<<grd, blk, 0, stream>>>(Wk, WkT, n, Dd);
        wT_to_bf16_kernel<<<grd, blk, 0, stream>>>(Wv, WvT, n, Dd);
        wT_to_bf16_kernel<<<grd, blk, 0, stream>>>(W1, W1T, n, Dd);
        wT_to_bf16_kernel<<<grd, blk, 0, stream>>>(W2, W2T, n, Dd);
    }

    // ---- H = x + wpe
    {
        int n = (int)HD, blk = 256, grd = (n + blk - 1) / blk;
        add_pe_kernel<<<grd, blk, 0, stream>>>(x, wpe, H, Hbf, n, Nn * Dd);
    }

    const dim3 gemmBlk(128);
    for (int l = 0; l < Ll; ++l) {
        const size_t wOff = (size_t)l * Dd * Dd;
        const size_t vOff = (size_t)l * Dd;

        // Q, K projections: (BN x D) * (D x D) -> bf16 row-major
        {
            dim3 grd(Dd / BNT, BN / BM);
            gemm_bf16_wmma<<<grd, gemmBlk, 0, stream>>>(Hbf, WqT + wOff, nullptr, nullptr,
                nullptr, Qbf, BN, Dd, Dd, 0);
            gemm_bf16_wmma<<<grd, gemmBlk, 0, stream>>>(Hbf, WkT + wOff, nullptr, nullptr,
                nullptr, Kbf, BN, Dd, Dd, 0);
        }
        // V projection per batch, stored TRANSPOSED [D][N] so attn*V stages async
        for (int bidx = 0; bidx < Bb; ++bidx) {
            const size_t o = (size_t)bidx * Nn * Dd;
            dim3 grd(Dd / BNT, Nn / BM);
            gemm_bf16_wmma<<<grd, gemmBlk, 0, stream>>>(Hbf + o, WvT + wOff, nullptr, nullptr,
                nullptr, VtBf + o, Nn, Dd, Dd, GF_STORE_BFT);
        }

        // attention per batch: S = relu(Q K^T)*mask (bf16), then Tmp = H + S*V
        for (int bidx = 0; bidx < Bb; ++bidx) {
            const size_t o = (size_t)bidx * Nn * Dd;
            {
                dim3 grd(Nn / BNT, Nn / BM);
                gemm_bf16_wmma<<<grd, gemmBlk, 0, stream>>>(Qbf + o, Kbf + o, nullptr, nullptr,
                    nullptr, Sbf, Nn, Nn, Dd, GF_RELU | GF_MASK);
            }
            {
                dim3 grd(Dd / BNT, Nn / BM);
                gemm_bf16_wmma<<<grd, gemmBlk, 0, stream>>>(Sbf, VtBf + o, nullptr, H + o,
                    Tmp + o, nullptr, Nn, Dd, Nn, GF_RESIDUAL | GF_CAUSALK);
            }
        }

        // LN1
        ln_kernel<<<BN, 256, 0, stream>>>(Tmp, ln1_g + vOff, ln1_b + vOff, H, Hbf, Dd);

        // MLP: h1 = relu(H*W1 + b1) (bf16); Tmp = H + h1*W2 + b2
        {
            dim3 grd(Dd / BNT, BN / BM);
            gemm_bf16_wmma<<<grd, gemmBlk, 0, stream>>>(Hbf, W1T + wOff, b1 + vOff, nullptr,
                nullptr, h1bf, BN, Dd, Dd, GF_BIAS | GF_RELU);
            gemm_bf16_wmma<<<grd, gemmBlk, 0, stream>>>(h1bf, W2T + wOff, b2 + vOff, H,
                Tmp, nullptr, BN, Dd, Dd, GF_BIAS | GF_RESIDUAL);
        }

        // LN2
        ln_kernel<<<BN, 256, 0, stream>>>(Tmp, ln2_g + vOff, ln2_b + vOff, H, Hbf, Dd);
    }

    // final H -> d_out
    {
        int n = (int)HD, blk = 256, grd = (n + blk - 1) / blk;
        copy_f32_kernel<<<grd, blk, 0, stream>>>(H, (float*)d_out, n);
    }
    (void)in_sizes; (void)n_in; (void)out_size; (void)ws_size;
}
